// GotenNet_73452530696703
// MI455X (gfx1250) — matrix operand order repaired and checked by reading, not compiled, not run on hardware
//
#include <hip/hip_runtime.h>
#include <hip/hip_bf16.h>
#include <float.h>
#include <math.h>
#include <stdint.h>

// Problem constants (B=1)
#define NT    128   // N tokens
#define DIM   64    // D
#define NH    8     // heads
#define DHD   128   // per-head dim
#define DI    1024  // NH*DHD
#define SDEG  5     // S
#define SDI   5120  // S*DI
#define DMLP  2048

typedef __attribute__((ext_vector_type(16))) __bf16 v16bf;
typedef __attribute__((ext_vector_type(8)))  float  v8f;

static __device__ __forceinline__ v8f zero8() {
  v8f z = {0.f, 0.f, 0.f, 0.f, 0.f, 0.f, 0.f, 0.f};
  return z;
}

// Async DMA of 16 bytes global -> LDS (CDNA5, tracked on ASYNCcnt).
static __device__ __forceinline__ void async_copy16(const float* gsrc, float* ldst) {
  unsigned loff = (unsigned)(uintptr_t)ldst;   // low 32 bits of generic LDS addr
  asm volatile("global_load_async_to_lds_b128 %0, %1, off"
               :: "v"(loff), "v"(gsrc) : "memory");
}
static __device__ __forceinline__ void async_wait0() {
  asm volatile("s_wait_asynccnt 0x0" ::: "memory");
}

// A fragment 16x32 (MxK) from row-major f32, lda in elements.
// lane l: m = l&15 ; K = kh*16 + e (e = 0..15), kh = l>>4
static __device__ __forceinline__ v16bf load_a_frag(const float* A, int lda, int lane) {
  const int m  = lane & 15;
  const int kh = (lane >> 4) & 1;
  const float* p0 = A + m * lda + kh * 16;
  v16bf a;
#pragma unroll
  for (int w = 0; w < 8; ++w)  a[w]     = (__bf16)p0[w];
#pragma unroll
  for (int w = 0; w < 8; ++w)  a[8 + w] = (__bf16)p0[8 + w];
  return a;
}

// B fragment 32x16 (KxN) from row-major f32, ldb in elements.
// lane l: n = l&15 ; K = kh*16 + e (e = 0..15), kh = l>>4
static __device__ __forceinline__ v16bf load_b_frag(const float* B, int ldb, int lane) {
  const int n  = lane & 15;
  const int kh = (lane >> 4) & 1;
  const float* p = B + (kh * 16) * ldb + n;
  v16bf b;
#pragma unroll
  for (int e = 0; e < 16; ++e) b[e] = (__bf16)p[e * ldb];
  return b;
}

static __device__ __forceinline__ v8f wmma_bf16(v16bf a, v16bf b, v8f c) {
  return __builtin_amdgcn_wmma_f32_16x16x32_bf16(false, a, false, b, (short)0, c,
                                                 false, false);
}

// C/D 16x16 f32: lane l: n = l&15 ; element r -> M = r + 8*(l>>4)
static __device__ __forceinline__ void store_c(float* C, int ldc, int lane, v8f c) {
  const int n  = lane & 15;
  const int mh = (lane >> 4) & 1;
#pragma unroll
  for (int r = 0; r < 8; ++r) C[(r + 8 * mh) * ldc + n] = c[r];
}

// ---------------- Kernel 1: LayerNorm (biased var, eps 1e-5) ----------------
__global__ __launch_bounds__(64) void k_layernorm(
    const float* __restrict__ h, const float* __restrict__ g_hi,
    const float* __restrict__ g_hj, float* __restrict__ hi, float* __restrict__ hj) {
  __shared__ float red[64];
  const int n = blockIdx.x, d = threadIdx.x;
  float x = h[n * DIM + d];
  red[d] = x; __syncthreads();
  for (int o = 32; o > 0; o >>= 1) { if (d < o) red[d] += red[d + o]; __syncthreads(); }
  float mu = red[0] * (1.0f / DIM);
  __syncthreads();
  float xc = x - mu;
  red[d] = xc * xc; __syncthreads();
  for (int o = 32; o > 0; o >>= 1) { if (d < o) red[d] += red[d + o]; __syncthreads(); }
  float r = rsqrtf(red[0] * (1.0f / DIM) + 1e-5f);
  hi[n * DIM + d] = xc * r * g_hi[d];
  hj[n * DIM + d] = xc * r * g_hj[d];
}

// ---------------- Kernel 2: q = hi@Wq, k = hj@Wk (128x64 @ 64x1024) ----------
__global__ __launch_bounds__(32) void k_gemm_qk(
    const float* __restrict__ hi, const float* __restrict__ hj,
    const float* __restrict__ Wq, const float* __restrict__ Wk,
    float* __restrict__ q, float* __restrict__ k) {
  const int lane = threadIdx.x;
  const int n0 = blockIdx.x * 16, m0 = blockIdx.y * 16, z = blockIdx.z;
  const float* A  = (z == 0) ? hi : hj;
  const float* Bm = (z == 0) ? Wq : Wk;
  float* C = (z == 0) ? q : k;
  v8f c = zero8();
#pragma unroll
  for (int kk = 0; kk < DIM; kk += 32) {
    v16bf a = load_a_frag(A + m0 * DIM + kk, DIM, lane);
    v16bf b = load_b_frag(Bm + kk * DI + n0, DI, lane);
    c = wmma_bf16(a, b, c);
  }
  store_c(C + m0 * DI + n0, DI, lane, c);
}

// ---------------- Kernel 3: t = silu(hj@W1 + b1)  (128x64 @ 64x2048) --------
__global__ __launch_bounds__(32) void k_mlp1(
    const float* __restrict__ hj, const float* __restrict__ Wv1,
    const float* __restrict__ bv1, const float* __restrict__ Wp1,
    const float* __restrict__ bp1, float* __restrict__ t1, float* __restrict__ t2) {
  const int lane = threadIdx.x;
  const int n0 = blockIdx.x * 16, m0 = blockIdx.y * 16, z = blockIdx.z;
  const float* Bm   = (z == 0) ? Wv1 : Wp1;
  const float* bias = (z == 0) ? bv1 : bp1;
  float* C = (z == 0) ? t1 : t2;
  v8f c = zero8();
#pragma unroll
  for (int kk = 0; kk < DIM; kk += 32) {
    v16bf a = load_a_frag(hj + m0 * DIM + kk, DIM, lane);
    v16bf b = load_b_frag(Bm + kk * DMLP + n0, DMLP, lane);
    c = wmma_bf16(a, b, c);
  }
  const int n = lane & 15, mh = (lane >> 4) & 1;
  const float bc = bias[n0 + n];
#pragma unroll
  for (int r = 0; r < 8; ++r) {
    float x = c[r] + bc;
    x = x / (1.0f + expf(-x));   // silu
    C[(m0 + r + 8 * mh) * DMLP + n0 + n] = x;
  }
}

// ---------------- Kernel 4: v = t1@Wv2 + bv2 (128x2048 @ 2048x5120) ---------
// block 256 (8 waves): 16 m-rows x 128 n-cols; A k-panel async-staged in LDS.
__global__ __launch_bounds__(256) void k_mlp2(
    const float* __restrict__ t1, const float* __restrict__ t2,
    const float* __restrict__ Wv2, const float* __restrict__ bv2,
    const float* __restrict__ Wp2, const float* __restrict__ bp2,
    float* __restrict__ v, float* __restrict__ pv) {
  __shared__ alignas(16) float sA[16 * 32];   // A panel for current k-step
  const int tid = threadIdx.x, lane = tid & 31, wave = tid >> 5;
  const int n0 = blockIdx.x * 128 + wave * 16;
  const int m0 = blockIdx.y * 16, z = blockIdx.z;
  const float* A    = (z == 0) ? t1 : t2;
  const float* Bm   = (z == 0) ? Wv2 : Wp2;
  const float* bias = (z == 0) ? bv2 : bp2;
  float* C = (z == 0) ? v : pv;
  v8f c = zero8();
  for (int kk = 0; kk < DMLP; kk += 32) {
    if (tid < 128) {   // 128 x 16B = 16x32 f32 panel, DMA'd straight to LDS
      const int row = tid >> 3, seg = tid & 7;
      async_copy16(A + (m0 + row) * DMLP + kk + seg * 4, &sA[row * 32 + seg * 4]);
    }
    if (kk + 32 < DMLP)   // prefetch next B panel rows for this wave's column strip
      __builtin_prefetch(Bm + (kk + 32 + (lane & 15)) * SDI + n0, 0, 1);
    async_wait0();
    __syncthreads();
    v16bf a = load_a_frag(sA, 32, lane);
    v16bf b = load_b_frag(Bm + kk * SDI + n0, SDI, lane);
    c = wmma_bf16(a, b, c);
    __syncthreads();
  }
  const int n = lane & 15, mh = (lane >> 4) & 1;
  const float bc = bias[n0 + n];
#pragma unroll
  for (int r = 0; r < 8; ++r)
    C[(m0 + r + 8 * mh) * SDI + n0 + n] = c[r] + bc;
}

// ---------------- Kernel 5: sim[h,i,s,j] = sum_dh q*k*silu(t_ij@Wek) --------
// grid: (jt=8, i=128, h=8), block 256 (8 waves)
__global__ __launch_bounds__(256) void k_sim(
    const float* __restrict__ t_ij, const float* __restrict__ Wek,
    const float* __restrict__ q, const float* __restrict__ kmat,
    float* __restrict__ sim) {
  __shared__ alignas(16) float ek[16 * 640];   // 40 KB: 16 j x (5 s * 128 dh)
  __shared__ alignas(16) float sAt[16 * 64];   // t_ij tile (async-staged)
  __shared__ float qrow[DHD];
  __shared__ float part[160];
  const int tid = threadIdx.x, lane = tid & 31, wave = tid >> 5;
  const int jt = blockIdx.x, i = blockIdx.y, hh = blockIdx.z;
  const int j0 = jt * 16;
  for (int t = tid; t < DHD; t += 256) qrow[t] = q[i * DI + hh * DHD + t];
  { // async-stage A tile: 256 threads x 16B = 16x64 f32
    const int row = tid >> 4, seg = tid & 15;
    async_copy16(t_ij + (i * NT + j0) * DIM + row * DIM + seg * 4,
                 &sAt[row * 64 + seg * 4]);
    async_wait0();
  }
  __syncthreads();
  // WMMA: ek_tile = sAt(16x64) @ Wek[:, cols(h)]  (16x640)
  for (int nt = wave * 5; nt < wave * 5 + 5; ++nt) {
    const int s = (nt * 16) / DHD, w = (nt * 16) % DHD;
    const float* Bm = Wek + s * DI + hh * DHD + w;
    v8f c = zero8();
#pragma unroll
    for (int kk = 0; kk < DIM; kk += 32) {
      v16bf a = load_a_frag(sAt + kk, DIM, lane);
      v16bf b = load_b_frag(Bm + kk * SDI, SDI, lane);
      c = wmma_bf16(a, b, c);
    }
    store_c(ek + nt * 16, 640, lane, c);
  }
  __syncthreads();
  for (int e = tid; e < 16 * 640; e += 256) {   // silu epilogue
    float x = ek[e];
    ek[e] = x / (1.0f + expf(-x));
  }
  __syncthreads();
  if (tid < 160) {     // (j,s) split 2-way over dh
    const int j = tid / 10, s = (tid % 10) >> 1, half = tid & 1;
    const float* kr = kmat + (j0 + j) * DI + hh * DHD + half * 64;
    const float* er = ek + j * 640 + s * DHD + half * 64;
    const float* qr = qrow + half * 64;
    float sum = 0.f;
#pragma unroll 8
    for (int dh = 0; dh < 64; ++dh) sum += qr[dh] * kr[dh] * er[dh];
    part[tid] = sum;
  }
  __syncthreads();
  if (tid < 80) {
    const int j = tid / SDEG, s = tid % SDEG;
    sim[((hh * NT + i) * SDEG + s) * NT + j0 + j] =
        part[j * 10 + s * 2] + part[j * 10 + s * 2 + 1];
  }
}

// ---------------- Kernel 6: masked softmax over j (rows of 128) -------------
__global__ __launch_bounds__(128) void k_softmax(float* __restrict__ sim,
                                                 const unsigned char* __restrict__ mask) {
  __shared__ float red[128];
  const int row = blockIdx.x, j = threadIdx.x;
  float* p = sim + row * NT;
  float x = mask[j] ? p[j] : -FLT_MAX;
  red[j] = x; __syncthreads();
  for (int o = 64; o > 0; o >>= 1) { if (j < o) red[j] = fmaxf(red[j], red[j + o]); __syncthreads(); }
  const float mx = red[0];
  __syncthreads();
  const float e = expf(x - mx);
  red[j] = e; __syncthreads();
  for (int o = 64; o > 0; o >>= 1) { if (j < o) red[j] += red[j + o]; __syncthreads(); }
  p[j] = e / red[0];
}

// ---------------- Kernel 7: fused ev + sea + Wo + output reductions ---------
// one block per query i; loops j-tiles(8) x heads(8)
__global__ __launch_bounds__(256) void k_out(
    const float* __restrict__ t_ij, const float* __restrict__ Wev,
    const float* __restrict__ Wo,   const float* __restrict__ attn,
    const float* __restrict__ vmat, const float* __restrict__ pvmat,
    const float* __restrict__ hin,  const float* __restrict__ x1,
    const float* __restrict__ x2,   const float* __restrict__ r1,
    const float* __restrict__ r2,   float* __restrict__ out) {
  __shared__ alignas(16) float sea[16 * 640];  // 40 KB; reused as out_tile (80x64)
  __shared__ alignas(16) float sAt[16 * 64];   // t_ij tile (async-staged per j-tile)
  __shared__ float acc[576];
  const int tid = threadIdx.x, lane = tid & 31, wave = tid >> 5;
  const int i = blockIdx.x;
  for (int e = tid; e < 576; e += 256) acc[e] = (e < 64) ? hin[i * DIM + e] : 0.f;
  __syncthreads();

  for (int jt = 0; jt < 8; ++jt) {
    const int j0 = jt * 16;
    { // async-stage the t_ij tile once per j-tile (reused across 8 heads)
      const int row = tid >> 4, seg = tid & 15;
      async_copy16(t_ij + (i * NT + j0) * DIM + row * DIM + seg * 4,
                   &sAt[row * 64 + seg * 4]);
      async_wait0();
    }
    __syncthreads();
    v8f oacc[3] = {zero8(), zero8(), zero8()};   // out_tile accumulators across heads

    for (int hh = 0; hh < NH; ++hh) {
      // GEMM1: ev_tile[16][640] = sAt(16x64) @ Wev[:, cols(hh)]
      for (int nt = wave * 5; nt < wave * 5 + 5; ++nt) {
        const int s = (nt * 16) / DHD, w = (nt * 16) % DHD;
        const float* Bm = Wev + s * DI + hh * DHD + w;
        v8f c = zero8();
#pragma unroll
        for (int kk = 0; kk < DIM; kk += 32) {
          v16bf a = load_a_frag(sAt + kk, DIM, lane);
          v16bf b = load_b_frag(Bm + kk * SDI, SDI, lane);
          c = wmma_bf16(a, b, c);
        }
        store_c(sea + nt * 16, 640, lane, c);
      }
      __syncthreads();
      // sea = attn * v + ev * pv   (elementwise)
      for (int e = tid; e < 16 * 640; e += 256) {
        const int j = e / 640, cix = e % 640, s = cix >> 7, dh = cix & 127;
        const int col = s * DI + hh * DHD + dh;
        const float ev = sea[e];
        const float at = attn[((hh * NT + i) * SDEG + s) * NT + j0 + j];
        sea[e] = at * vmat[(j0 + j) * SDI + col] + ev * pvmat[(j0 + j) * SDI + col];
      }
      __syncthreads();
      // GEMM2: out_tile[80][64] += sea(80x128) @ Wo[hh*128:(hh+1)*128, :]
      for (int u = 0; u < 3; ++u) {
        const int t = wave + u * 8;
        if (t < 20) {
          const int mt = t >> 2, nt2 = t & 3;
          const float* Ap = sea + mt * 16 * DHD;
          const float* Bp = Wo + (hh * DHD) * DIM + nt2 * 16;
#pragma unroll
          for (int kk = 0; kk < DHD; kk += 32) {
            v16bf a = load_a_frag(Ap + kk, DHD, lane);
            v16bf b = load_b_frag(Bp + kk * DIM, DIM, lane);
            oacc[u] = wmma_bf16(a, b, oacc[u]);
          }
        }
      }
      __syncthreads();
    }

    // dump out_tile (rows r = j*5+s, cols d) into LDS (reuse sea)
    for (int u = 0; u < 3; ++u) {
      const int t = wave + u * 8;
      if (t < 20) {
        const int mt = t >> 2, nt2 = t & 3;
        store_c(sea + (mt * 16) * DIM + nt2 * 16, DIM, lane, oacc[u]);
      }
    }
    __syncthreads();
    // per-j-tile reductions into the 576-entry per-i accumulator
    for (int e = tid; e < 576; e += 256) {
      float sum = 0.f;
      if (e < 64) {
        const int d = e;
        for (int j = 0; j < 16; ++j) sum += sea[(j * SDEG + 0) * DIM + d];
      } else if (e < 256) {
        const int d = (e - 64) / 3, m = (e - 64) % 3;
        for (int j = 0; j < 16; ++j) {
          sum += x1[(j0 + j) * (DIM * 3) + d * 3 + m] * sea[(j * SDEG + 3) * DIM + d];
          sum += r1[(i * NT + j0 + j) * 3 + m]        * sea[(j * SDEG + 1) * DIM + d];
        }
      } else {
        const int d = (e - 256) / 5, m = (e - 256) % 5;
        for (int j = 0; j < 16; ++j) {
          sum += x2[(j0 + j) * (DIM * 5) + d * 5 + m] * sea[(j * SDEG + 4) * DIM + d];
          sum += r2[(i * NT + j0 + j) * 5 + m]        * sea[(j * SDEG + 2) * DIM + d];
        }
      }
      acc[e] += sum;
    }
    __syncthreads();
  }
  for (int e = tid; e < 576; e += 256) out[i * 576 + e] = acc[e];
}

// ---------------------------------------------------------------------------
extern "C" void kernel_launch(void* const* d_in, const int* in_sizes, int n_in,
                              void* d_out, int out_size, void* d_ws, size_t ws_size,
                              hipStream_t stream) {
  (void)in_sizes; (void)n_in; (void)out_size; (void)ws_size;
  const float* h    = (const float*)d_in[0];
  const float* t_ij = (const float*)d_in[1];
  const float* r1   = (const float*)d_in[2];
  const float* r2   = (const float*)d_in[3];
  const float* x1   = (const float*)d_in[4];
  const float* x2   = (const float*)d_in[5];
  const float* g_hi = (const float*)d_in[6];
  const float* g_hj = (const float*)d_in[7];
  const float* Wq   = (const float*)d_in[8];
  const float* Wk   = (const float*)d_in[9];
  const float* Wv1  = (const float*)d_in[10];
  const float* bv1  = (const float*)d_in[11];
  const float* Wv2  = (const float*)d_in[12];
  const float* bv2  = (const float*)d_in[13];
  const float* Wek  = (const float*)d_in[14];
  const float* Wev  = (const float*)d_in[15];
  const float* Wp1  = (const float*)d_in[16];
  const float* bp1  = (const float*)d_in[17];
  const float* Wp2  = (const float*)d_in[18];
  const float* bp2  = (const float*)d_in[19];
  const float* Wo   = (const float*)d_in[20];
  const unsigned char* mask = (const unsigned char*)d_in[21];
  float* out = (float*)d_out;

  float* ws   = (float*)d_ws;                 // workspace layout (floats)
  float* hi   = ws;                           //   8,192
  float* hjv  = ws + 8192;                    //   8,192
  float* qb   = ws + 16384;                   // 131,072
  float* kb   = ws + 147456;                  // 131,072
  float* t1   = ws + 278528;                  // 262,144
  float* t2   = ws + 540672;                  // 262,144
  float* vb   = ws + 802816;                  // 655,360
  float* pvb  = ws + 1458176;                 // 655,360
  float* simb = ws + 2113536;                 // 655,360  (attn in-place) -> 2,768,896 total

  hipLaunchKernelGGL(k_layernorm, dim3(NT), dim3(64), 0, stream, h, g_hi, g_hj, hi, hjv);
  hipLaunchKernelGGL(k_gemm_qk, dim3(DI / 16, NT / 16, 2), dim3(32), 0, stream,
                     hi, hjv, Wq, Wk, qb, kb);
  hipLaunchKernelGGL(k_mlp1, dim3(DMLP / 16, NT / 16, 2), dim3(32), 0, stream,
                     hjv, Wv1, bv1, Wp1, bp1, t1, t2);
  hipLaunchKernelGGL(k_mlp2, dim3(SDI / 128, NT / 16, 2), dim3(256), 0, stream,
                     t1, t2, Wv2, bv2, Wp2, bp2, vb, pvb);
  hipLaunchKernelGGL(k_sim, dim3(NT / 16, NT, NH), dim3(256), 0, stream,
                     t_ij, Wek, qb, kb, simb);
  hipLaunchKernelGGL(k_softmax, dim3(NH * NT * SDEG), dim3(128), 0, stream, simb, mask);
  hipLaunchKernelGGL(k_out, dim3(NT), dim3(256), 0, stream,
                     t_ij, Wev, Wo, simb, vb, pvb, h, x1, x2, r1, r2, out);
}
